// ResidualBlockOriginal_40699110096963
// MI455X (gfx1250) — matrix operand order repaired and checked by reading, not compiled, not run on hardware
//
#include <hip/hip_runtime.h>

// ---------------------------------------------------------------------------
// CDNA5 (gfx1250) sparse-conv residual block using v_wmma_f32_16x16x32_bf16.
// ---------------------------------------------------------------------------

typedef __attribute__((ext_vector_type(16))) __bf16 v16bf;
typedef __attribute__((ext_vector_type(8)))  __bf16 v8bf;
typedef __attribute__((ext_vector_type(8)))  float  v8f;
typedef __attribute__((ext_vector_type(4)))  float  v4f;

#define C_CH 32
#define K_OFF 27
#define EPSV 1e-5f

// ---------------------------------------------------------------------------
// Repack per-offset 32x32 f32 weights into per-lane bf16 B-fragments for
// v_wmma_f32_16x16x32_bf16.  Fragment layout (ISA 7.12.2 style):
//   lane L: col n = L&15, ksel = L>>4 ; element j: K = (j<8 ? ksel*8 + j : 16 + ksel*8 + j-8), value W[K][col]
// Stored as frag[k][h][lane][16] contiguous -> one 32B coalesced load per lane.
// grid: K_OFF blocks x 64 threads (h = tid>>5, lane = tid&31)
// ---------------------------------------------------------------------------
__global__ void prep_wfrag_kernel(const float* __restrict__ W,
                                  __bf16* __restrict__ frag) {
    const int k    = blockIdx.x;
    const int tid  = threadIdx.x;
    const int h    = tid >> 5;        // cout half: 0 or 1
    const int lane = tid & 31;
    const int n    = lane & 15;
    const int ksel = lane >> 4;
    const float* Wk = W + k * C_CH * C_CH;
    __bf16* out = frag + (((long)k * 2 + h) * 32 + lane) * 16;
#pragma unroll
    for (int j = 0; j < 16; ++j) {
        int kk = (j < 8) ? (ksel * 8 + j) : (16 + ksel * 8 + (j - 8));
        out[j] = (__bf16)Wk[kk * C_CH + h * 16 + n];
    }
}

// ---------------------------------------------------------------------------
// Sparse convolution: one wave per (offset k, 16-row tile).
//   A (16x32, bf16): lane L holds row m=L&15, channels ksel*8+{0..7} (elems 0-7)
//                    and 16+ksel*8+{0..7} (elems 8-15), per ISA 7.12.2.
//   Two WMMAs cover cout halves 0-15 / 16-31.
//   D layout (documented): lane L, vgpr v -> (M = (L>>4)*8 + v, N = L&15).
//   Scatter-add via native f32 global atomics through out_map; rows whose mask
//   is 0 contribute exact zeros and are skipped entirely (validity + out_map
//   distributed by wave32 shuffle from the coalesced gather-phase loads).
// ---------------------------------------------------------------------------
template <bool IN_BF16>
__global__ __launch_bounds__(256)
void spconv_wmma_kernel(const void* __restrict__ feats,        // [N,32] f32 or bf16
                        const __bf16* __restrict__ wfrag,      // [K,2,32,16]
                        const int* __restrict__ in_map,        // [K,N]
                        const int* __restrict__ out_map,       // [K,N]
                        const unsigned char* __restrict__ mask,// [K,N] (bool bytes)
                        float* __restrict__ acc,               // [N,32] accumulator
                        int N) {
    const int k    = blockIdx.y;
    const int wave = threadIdx.x >> 5;
    const int lane = threadIdx.x & 31;
    const int tile = blockIdx.x * (blockDim.x >> 5) + wave;
    const int numTiles = (N + 15) >> 4;
    if (tile >= numTiles) return;          // wave-uniform: EXEC stays all-ones

    const int  ksel    = lane >> 4;
    const int  m       = lane & 15;
    const int  row_idx = tile * 16 + m;
    const long kbase   = (long)k * N;

    // ---- coalesced map/mask loads for this lane's gather row ----
    int  src    = 0;
    int  orow   = 0;
    int  validi = 0;
    if (row_idx < N) {
        src    = in_map[kbase + row_idx];
        orow   = out_map[kbase + row_idx];
        validi = (mask[kbase + row_idx] != 0) ? 1 : 0;
    }

    // ---- A fragment: masked gather + cvt to bf16 ----
    v16bf a;
    const int c0 = ksel * 8;
    if (validi) {
        if (IN_BF16) {
            const __bf16* f = (const __bf16*)feats + (long)src * C_CH;
            v8bf lo = *(const v8bf*)(f + c0);
            v8bf hi = *(const v8bf*)(f + 16 + c0);
#pragma unroll
            for (int j = 0; j < 8; ++j) { a[j] = lo[j]; a[8 + j] = hi[j]; }
        } else {
            const float* f = (const float*)feats + (long)src * C_CH;
            v4f l0 = *(const v4f*)(f + c0);
            v4f l1 = *(const v4f*)(f + c0 + 4);
            v4f h0 = *(const v4f*)(f + 16 + c0);
            v4f h1 = *(const v4f*)(f + 16 + c0 + 4);
#pragma unroll
            for (int j = 0; j < 4; ++j) {
                a[j]      = (__bf16)l0[j];
                a[4 + j]  = (__bf16)l1[j];
                a[8 + j]  = (__bf16)h0[j];
                a[12 + j] = (__bf16)h1[j];
            }
        }
    } else {
#pragma unroll
        for (int j = 0; j < 16; ++j) a[j] = (__bf16)0.0f;
    }

    // ---- B fragments: coalesced 32B load per lane ----
    const __bf16* wf = wfrag + (long)k * 2 * 32 * 16 + lane * 16;
    v16bf b0 = *(const v16bf*)(wf);
    v16bf b1 = *(const v16bf*)(wf + 32 * 16);

    // ---- 16x16x32 bf16 WMMAs, f32 accumulate ----
    v8f c0v = {};
    v8f c1v = {};
    c0v = __builtin_amdgcn_wmma_f32_16x16x32_bf16(false, a, false, b0,
                                                  (short)0, c0v, false, false);
    c1v = __builtin_amdgcn_wmma_f32_16x16x32_bf16(false, a, false, b1,
                                                  (short)0, c1v, false, false);

    // ---- scatter-add: skip exact-zero (masked) rows ----
#pragma unroll
    for (int v = 0; v < 8; ++v) {
        const int srcl = ksel * 8 + v;              // lane that gathered row M
        const int ov   = __shfl(orow, srcl, 32);
        const int mv   = __shfl(validi, srcl, 32);
        if (mv) {
            float* p = acc + (long)ov * C_CH + m;
            unsafeAtomicAdd(p,      c0v[v]);
            unsafeAtomicAdd(p + 16, c1v[v]);
        }
    }
}

// ---------------------------------------------------------------------------
// Per-channel sum / sum-of-squares over N rows.
// stats[0..31] = sum, stats[32..63] = sumsq.
// ---------------------------------------------------------------------------
__global__ __launch_bounds__(256)
void bn_stats_kernel(const float* __restrict__ y, float* __restrict__ stats, int N) {
    __shared__ float s_sum[256];
    __shared__ float s_sq[256];
    const int c    = threadIdx.x & 31;
    const int rgrp = threadIdx.x >> 5;     // 0..7 row groups
    float s = 0.0f, s2 = 0.0f;
    for (long row = (long)blockIdx.x * 8 + rgrp; row < N; row += (long)gridDim.x * 8) {
        float v = y[row * C_CH + c];
        s  += v;
        s2 += v * v;
    }
    s_sum[threadIdx.x] = s;
    s_sq[threadIdx.x]  = s2;
    __syncthreads();
    if (rgrp == 0) {
#pragma unroll
        for (int i = 1; i < 8; ++i) {
            s  += s_sum[c + i * 32];
            s2 += s_sq[c + i * 32];
        }
        unsafeAtomicAdd(&stats[c], s);
        unsafeAtomicAdd(&stats[32 + c], s2);
    }
}

// ---------------------------------------------------------------------------
// y1n = bf16( relu( (y - mean) * gamma / sqrt(var+eps) + beta ) )
// ---------------------------------------------------------------------------
__global__ __launch_bounds__(256)
void bn_relu_to_bf16_kernel(const float* __restrict__ y,
                            const float* __restrict__ stats,
                            const float* __restrict__ gamma,
                            const float* __restrict__ beta,
                            __bf16* __restrict__ out, int N) {
    const long  total = (long)N * C_CH;
    const float invN  = 1.0f / (float)N;
    for (long i = (long)blockIdx.x * blockDim.x + threadIdx.x; i < total;
         i += (long)gridDim.x * blockDim.x) {
        const int c   = (int)(i & (C_CH - 1));
        float mean    = stats[c] * invN;
        float var     = stats[32 + c] * invN - mean * mean;
        float sc      = gamma[c] * rsqrtf(var + EPSV);
        float sh      = beta[c] - mean * sc;
        float v       = fmaf(y[i], sc, sh);
        v             = v > 0.0f ? v : 0.0f;
        out[i]        = (__bf16)v;
    }
}

// ---------------------------------------------------------------------------
// out = relu( bn2(y) + x )
// ---------------------------------------------------------------------------
__global__ __launch_bounds__(256)
void bn_add_relu_kernel(const float* __restrict__ y,
                        const float* __restrict__ stats,
                        const float* __restrict__ gamma,
                        const float* __restrict__ beta,
                        const float* __restrict__ x,
                        float* __restrict__ out, int N) {
    const long  total = (long)N * C_CH;
    const float invN  = 1.0f / (float)N;
    for (long i = (long)blockIdx.x * blockDim.x + threadIdx.x; i < total;
         i += (long)gridDim.x * blockDim.x) {
        const int c   = (int)(i & (C_CH - 1));
        float mean    = stats[c] * invN;
        float var     = stats[32 + c] * invN - mean * mean;
        float sc      = gamma[c] * rsqrtf(var + EPSV);
        float sh      = beta[c] - mean * sc;
        float v       = fmaf(y[i], sc, sh) + x[i];
        out[i]        = v > 0.0f ? v : 0.0f;
    }
}

// ---------------------------------------------------------------------------
// Launch: conv1 -> bn1 stats -> bn1+relu (bf16) -> conv2 -> bn2 stats -> fuse
// ---------------------------------------------------------------------------
extern "C" void kernel_launch(void* const* d_in, const int* in_sizes, int n_in,
                              void* d_out, int out_size, void* d_ws, size_t ws_size,
                              hipStream_t stream) {
    const float* x      = (const float*)d_in[0];
    const float* W1     = (const float*)d_in[1];
    const float* gamma1 = (const float*)d_in[2];
    const float* beta1  = (const float*)d_in[3];
    const float* W2     = (const float*)d_in[4];
    const float* gamma2 = (const float*)d_in[5];
    const float* beta2  = (const float*)d_in[6];
    const int*   in_map = (const int*)d_in[7];
    const int*   out_map= (const int*)d_in[8];
    const unsigned char* mask = (const unsigned char*)d_in[9];
    float* out = (float*)d_out;

    const int N = in_sizes[0] / C_CH;

    // ---- workspace carve-up ----
    float*  yacc   = (float*)d_ws;                          // [N,32] f32 accumulator (reused)
    __bf16* y1n    = (__bf16*)(yacc + (long)N * C_CH);      // [N,32] bf16 intermediate
    __bf16* wf1    = y1n + (long)N * C_CH;                  // [27,2,32,16]
    __bf16* wf2    = wf1 + (long)K_OFF * 2 * 32 * 16;
    float*  stats1 = (float*)(wf2 + (long)K_OFF * 2 * 32 * 16); // 64 f32
    float*  stats2 = stats1 + 64;                               // 64 f32

    const int numTiles = (N + 15) >> 4;
    const int wavesPerBlock = 8;                            // 256 threads
    dim3 convGrid((numTiles + wavesPerBlock - 1) / wavesPerBlock, K_OFF);
    dim3 convBlock(32 * wavesPerBlock);
    const int statGrid = 1024;
    const int ewGrid   = 2048;

    // zero accumulator + stats
    hipMemsetAsync(yacc, 0, (size_t)N * C_CH * sizeof(float), stream);
    hipMemsetAsync(stats1, 0, 2 * 64 * sizeof(float), stream);

    // weight fragment repack (both convs)
    prep_wfrag_kernel<<<K_OFF, 64, 0, stream>>>(W1, wf1);
    prep_wfrag_kernel<<<K_OFF, 64, 0, stream>>>(W2, wf2);

    // conv1 (f32 input) -> yacc
    spconv_wmma_kernel<false><<<convGrid, convBlock, 0, stream>>>(
        (const void*)x, wf1, in_map, out_map, mask, yacc, N);

    // bn1 stats + bn1/relu -> bf16
    bn_stats_kernel<<<statGrid, 256, 0, stream>>>(yacc, stats1, N);
    bn_relu_to_bf16_kernel<<<ewGrid, 256, 0, stream>>>(yacc, stats1, gamma1, beta1, y1n, N);

    // conv2 (bf16 input) -> yacc (re-zeroed)
    hipMemsetAsync(yacc, 0, (size_t)N * C_CH * sizeof(float), stream);
    spconv_wmma_kernel<true><<<convGrid, convBlock, 0, stream>>>(
        (const void*)y1n, wf2, in_map, out_map, mask, yacc, N);

    // bn2 stats + fused bn2 + residual + relu -> out
    bn_stats_kernel<<<statGrid, 256, 0, stream>>>(yacc, stats2, N);
    bn_add_relu_kernel<<<ewGrid, 256, 0, stream>>>(yacc, stats2, gamma2, beta2, x, out, N);
}
